// MLA_20899310862928
// MI455X (gfx1250) — matrix operand order repaired and checked by reading, not compiled, not run on hardware
//
#include <hip/hip_runtime.h>
#include <hip/hip_bf16.h>
#include <math.h>

#define BATCH 8
#define SEQ 1024
#define EMBED 2048
#define HEADS 16
#define LATENT 512
#define HEAD_DIM 128
#define NOPE 32

typedef __attribute__((ext_vector_type(16))) __bf16 v16bf;
typedef __attribute__((ext_vector_type(8)))  __bf16 v8bf;
typedef __attribute__((ext_vector_type(4)))  __bf16 v4bf;
typedef __attribute__((ext_vector_type(8)))  float  v8f;

static __device__ __forceinline__ v16bf make_frag(v8bf lo, v8bf hi) {
  v16bf f;
#pragma unroll
  for (int e = 0; e < 8; ++e) { f[e] = lo[e]; f[e + 8] = hi[e]; }
  return f;
}

// ---- CDNA5 async global->LDS copy (ASYNCcnt path, CDNA5 ISA + 07 §15.18.3)
static __device__ __forceinline__ uint32_t lds_off(const void* p) {
  return (uint32_t)(uintptr_t)p;  // low 32 bits of generic addr = LDS offset
}
static __device__ __forceinline__ void async_b128(uint32_t lds, const void* g) {
  asm volatile("global_load_async_to_lds_b128 %0, %1, off"
               :
               : "v"(lds), "v"((unsigned long long)(uintptr_t)g)
               : "memory");
}
static __device__ __forceinline__ void wait_async0() {
  asm volatile("s_wait_asynccnt 0x0" ::: "memory");
}

// ---------------------------------------------------------------- fp32 -> bf16
__global__ __launch_bounds__(256) void cvt_f32_bf16(const float* __restrict__ in,
                                                    __bf16* __restrict__ out,
                                                    int n4) {
  int i = blockIdx.x * blockDim.x + threadIdx.x;
  if (i < n4) {
    float4 f = ((const float4*)in)[i];
    v4bf o = {(__bf16)f.x, (__bf16)f.y, (__bf16)f.z, (__bf16)f.w};
    ((v4bf*)out)[i] = o;
  }
}

// ------------------------------------------------ fp32 -> bf16 with transpose
// in: K x N fp32 row-major;  out: N x K bf16 row-major (pre-transposed weight)
__global__ __launch_bounds__(256) void cvt_t_f32_bf16(const float* __restrict__ in,
                                                      __bf16* __restrict__ out,
                                                      int K, int N) {
  __shared__ float t[32][33];
  const int bk = blockIdx.x * 32, bn = blockIdx.y * 32;
  const int tx = threadIdx.x & 31, ty = threadIdx.x >> 5;  // ty: 0..7
#pragma unroll
  for (int i = 0; i < 32; i += 8)
    t[ty + i][tx] = in[(size_t)(bk + ty + i) * N + bn + tx];
  __syncthreads();
#pragma unroll
  for (int i = 0; i < 32; i += 8)
    out[(size_t)(bn + ty + i) * K + bk + tx] = (__bf16)t[tx][ty + i];
}

// ---------------------------------------------------------------- bf16 GEMM
// C[M,N] = A[M,K] @ Bt[N,K]^T, bf16 in, f32 accumulate via WMMA.
// 256 threads = 8 waves (4m x 2n); block tile 128x128; wave tile 32x64
// (8 WMMAs per k-chunk). K chunks of 32, LDS double-buffered with async
// global->LDS copies overlapped against the WMMAs.
#define GTM 128
#define GTN 128
#define GTK 32

template <bool STORE_T>
__global__ __launch_bounds__(256) void gemm_bf16_wmma(
    const __bf16* __restrict__ A, const __bf16* __restrict__ Bt,
    __bf16* __restrict__ C, int M, int N, int K) {
  __shared__ __bf16 As[2][GTM][GTK];   // 2 x 8 KB
  __shared__ __bf16 Bs[2][GTN][GTK];   // 2 x 8 KB ([n][k], pre-transposed W)
  const int tid  = threadIdx.x;
  const int wave = tid >> 5;
  const int wm   = wave >> 1;          // 0..3
  const int wn   = wave & 1;           // 0..1
  const int lane = tid & 31;
  const int half = lane >> 4;
  const int l16  = lane & 15;
  const int bm = blockIdx.x * GTM;
  const int bn = blockIdx.y * GTN;

  v8f acc[2][4];
#pragma unroll
  for (int am = 0; am < 2; ++am)
#pragma unroll
    for (int bt = 0; bt < 4; ++bt)
#pragma unroll
      for (int v = 0; v < 8; ++v) acc[am][bt][v] = 0.0f;

  // stage one 128x32 tile pair: 2 x 2 async b128 per thread, contiguous source
  auto stage = [&](int buf, int k0) {
#pragma unroll
    for (int u = 0; u < 2; ++u) {
      int uid = tid + u * 256;
      int r = uid >> 2;            // 0..127
      int c = (uid & 3) * 8;       // 0,8,16,24
      async_b128(lds_off(&As[buf][r][c]), &A[(size_t)(bm + r) * K + k0 + c]);
      async_b128(lds_off(&Bs[buf][r][c]), &Bt[(size_t)(bn + r) * K + k0 + c]);
    }
  };

  stage(0, 0);
  wait_async0();
  __syncthreads();

  int buf = 0;
  for (int k0 = 0; k0 < K; k0 += GTK) {
    if (k0 + GTK < K) stage(buf ^ 1, k0 + GTK);  // overlap copy with compute

    const int khalf = half * 8;
    v16bf afrag[2];
#pragma unroll
    for (int am = 0; am < 2; ++am) {
      const __bf16* ap = &As[buf][32 * wm + 16 * am + l16][khalf];
      afrag[am] = make_frag(*(const v8bf*)ap, *(const v8bf*)(ap + 16));
    }
#pragma unroll
    for (int bt = 0; bt < 4; ++bt) {
      const __bf16* bp = &Bs[buf][64 * wn + 16 * bt + l16][khalf];
      v16bf bfrag = make_frag(*(const v8bf*)bp, *(const v8bf*)(bp + 16));
#pragma unroll
      for (int am = 0; am < 2; ++am)
        acc[am][bt] = __builtin_amdgcn_wmma_f32_16x16x32_bf16(
            false, afrag[am], false, bfrag, (short)0, acc[am][bt], false, false);
    }

    wait_async0();
    __syncthreads();
    buf ^= 1;
  }

  // epilogue: C tile layout row = v + 8*half, col = l16
#pragma unroll
  for (int am = 0; am < 2; ++am)
#pragma unroll
    for (int bt = 0; bt < 4; ++bt)
#pragma unroll
      for (int v = 0; v < 8; ++v) {
        int r = bm + 32 * wm + 16 * am + v + 8 * half;
        int c = bn + 64 * wn + 16 * bt + l16;
        if (!STORE_T) {
          C[(size_t)r * N + c] = (__bf16)acc[am][bt][v];
        } else {
          // store transposed as (B, N, SEQ): V output, so attention can stage
          // [d][key] tiles with straight contiguous async copies
          int bb = r >> 10;        // r / SEQ
          int ss = r & 1023;       // r % SEQ
          C[((size_t)bb * N + c) * SEQ + ss] = (__bf16)acc[am][bt][v];
        }
      }
}

// ---------------------------------------------------------------- RoPE (in place)
__global__ __launch_bounds__(128) void rope_inplace(__bf16* __restrict__ X) {
  const int idx = blockIdx.x * blockDim.x + threadIdx.x;
  if (idx >= BATCH * SEQ * HEADS) return;
  const int h  = idx % HEADS;
  const int bs = idx / HEADS;
  const int s  = bs % SEQ;
  __bf16* p = X + (size_t)bs * EMBED + h * HEAD_DIM + NOPE;
  float xe[48], xo[48];
#pragma unroll
  for (int j = 0; j < 48; ++j) {
    xe[j] = (float)p[2 * j];
    xo[j] = (float)p[2 * j + 1];
  }
#pragma unroll
  for (int j = 0; j < 48; ++j) {
    const float freq = __expf(-9.210340371976184f * (float)(2 * j) / 96.0f);
    const float ang  = (float)s * freq;
    float c, sn;
    __sincosf(ang, &c, &sn);
    p[j]      = (__bf16)(xe[j] * c - xo[j] * sn);
    p[j + 48] = (__bf16)(xo[j] * c + xe[j] * sn);
  }
}

// ---------------------------------------------------------------- attention
// 256 threads (8 waves) per 128 query rows of one (b,h); wave w owns 16 rows.
// Flash loop over 64-key blocks, causal early exit, K/V tiles double-buffered
// with async global->LDS copies. K staged [key][d]; V read from the
// pre-transposed (B,E,S) buffer so [d][key] staging is a straight copy.
#define AQ 128
#define AK 64

__global__ __launch_bounds__(256) void mla_attention(
    const __bf16* __restrict__ Q, const __bf16* __restrict__ Kt,
    const __bf16* __restrict__ Vtg, float* __restrict__ O, float scale) {
  __shared__ __bf16 Ks[2][AK][HEAD_DIM];  // [key][d]   2 x 16 KB
  __shared__ __bf16 Vt[2][HEAD_DIM][AK];  // [d][key]   2 x 16 KB
  __shared__ __bf16 Pw[8][16][AK];        // per-wave P     16 KB

  const int tid  = threadIdx.x;
  const int wave = tid >> 5;
  const int lane = tid & 31;
  const int half = lane >> 4;
  const int l16  = lane & 15;

  const int b = blockIdx.x / HEADS;
  const int h = blockIdx.x % HEADS;
  const int qbase = blockIdx.y * AQ;
  const int qrow0 = qbase + 16 * wave;
  const size_t head_off = (size_t)h * HEAD_DIM;

  // Q fragments: 4 chunks of 32 along d, held in registers
  v16bf qf[4];
  {
    const size_t roff = (size_t)(b * SEQ + qrow0 + l16) * EMBED + head_off;
    const int khalf = half * 8;
#pragma unroll
    for (int kc = 0; kc < 4; ++kc) {
      const __bf16* p = &Q[roff + 32 * kc + khalf];
      qf[kc] = make_frag(*(const v8bf*)p, *(const v8bf*)(p + 16));
    }
  }

  float mrun[8], lrun[8];
  v8f oacc[8];
#pragma unroll
  for (int v = 0; v < 8; ++v) { mrun[v] = -3.0e38f; lrun[v] = 0.0f; }
#pragma unroll
  for (int t = 0; t < 8; ++t)
#pragma unroll
    for (int v = 0; v < 8; ++v) oacc[t][v] = 0.0f;

  // stage one K/V block: 4 + 4 async b128 per thread, contiguous sources
  auto stage = [&](int sb, int kb) {
#pragma unroll
    for (int u = 0; u < 4; ++u) {
      int uid = tid + u * 256;
      int r = uid >> 4;            // key 0..63
      int c = (uid & 15) * 8;      // d
      async_b128(lds_off(&Ks[sb][r][c]),
                 &Kt[(size_t)(b * SEQ + kb + r) * EMBED + head_off + c]);
    }
#pragma unroll
    for (int u = 0; u < 4; ++u) {
      int uid = tid + u * 256;
      int d = uid >> 3;            // 0..127
      int kc = (uid & 7) * 8;      // key chunk
      async_b128(lds_off(&Vt[sb][d][kc]),
                 &Vtg[((size_t)b * EMBED + head_off + d) * SEQ + kb + kc]);
    }
  };

  const int kend = qbase + AQ;
  stage(0, 0);
  wait_async0();
  __syncthreads();

  int buf = 0;
  for (int kb = 0; kb < kend; kb += AK) {
    if (kb + AK < kend) stage(buf ^ 1, kb + AK);  // overlap copy with compute

    // S = Q @ K^T : wave computes 16x64 (4 tiles), contraction over d
    v8f sacc[4];
#pragma unroll
    for (int t = 0; t < 4; ++t)
#pragma unroll
      for (int v = 0; v < 8; ++v) sacc[t][v] = 0.0f;
#pragma unroll
    for (int nk = 0; nk < 4; ++nk)
#pragma unroll
      for (int kc = 0; kc < 4; ++kc) {
        const __bf16* bp = &Ks[buf][16 * nk + l16][32 * kc + 8 * half];
        v16bf bfrag = make_frag(*(const v8bf*)bp, *(const v8bf*)(bp + 16));
        sacc[nk] = __builtin_amdgcn_wmma_f32_16x16x32_bf16(
            false, qf[kc], false, bfrag, (short)0, sacc[nk], false, false);
      }

    // causal mask + online softmax (row = v + 8*half; 16-lane xor reductions)
    float fcorr[8];
#pragma unroll
    for (int v = 0; v < 8; ++v) {
      const int qg = qrow0 + v + 8 * half;
      float rmax = -3.0e38f;
#pragma unroll
      for (int nk = 0; nk < 4; ++nk) {
        const int kg = kb + 16 * nk + l16;
        float s = (kg <= qg) ? sacc[nk][v] * scale : -3.0e38f;
        sacc[nk][v] = s;
        rmax = fmaxf(rmax, s);
      }
#pragma unroll
      for (int m = 1; m < 16; m <<= 1)
        rmax = fmaxf(rmax, __shfl_xor(rmax, m, 32));
      const float mnew = fmaxf(mrun[v], rmax);
      const float f = __expf(mrun[v] - mnew);
      float rsum = 0.0f;
#pragma unroll
      for (int nk = 0; nk < 4; ++nk) {
        float pv = __expf(sacc[nk][v] - mnew);
        sacc[nk][v] = pv;
        rsum += pv;
      }
#pragma unroll
      for (int m = 1; m < 16; m <<= 1)
        rsum += __shfl_xor(rsum, m, 32);
      lrun[v] = lrun[v] * f + rsum;
      mrun[v] = mnew;
      fcorr[v] = f;
    }
#pragma unroll
    for (int t = 0; t < 8; ++t)
#pragma unroll
      for (int v = 0; v < 8; ++v) oacc[t][v] *= fcorr[v];

    // C-layout -> A-layout reshape of P through per-wave LDS tile
#pragma unroll
    for (int nk = 0; nk < 4; ++nk)
#pragma unroll
      for (int v = 0; v < 8; ++v)
        Pw[wave][v + 8 * half][16 * nk + l16] = (__bf16)sacc[nk][v];

    // O += P @ V : contraction over 64 keys = 2 chunks of 32
#pragma unroll
    for (int kc2 = 0; kc2 < 2; ++kc2) {
      const __bf16* pp = &Pw[wave][l16][32 * kc2 + 8 * half];
      v16bf pfrag = make_frag(*(const v8bf*)pp, *(const v8bf*)(pp + 16));
#pragma unroll
      for (int t = 0; t < 8; ++t) {
        const __bf16* vp = &Vt[buf][16 * t + l16][32 * kc2 + 8 * half];
        v16bf vfrag = make_frag(*(const v8bf*)vp, *(const v8bf*)(vp + 16));
        oacc[t] = __builtin_amdgcn_wmma_f32_16x16x32_bf16(
            false, pfrag, false, vfrag, (short)0, oacc[t], false, false);
      }
    }

    wait_async0();
    __syncthreads();
    buf ^= 1;
  }

  // normalize, write fp32 (B,S,H*D)
#pragma unroll
  for (int v = 0; v < 8; ++v) {
    const int qg = qrow0 + v + 8 * half;
    const float inv = 1.0f / lrun[v];
    const size_t base = (size_t)(b * SEQ + qg) * EMBED + head_off;
#pragma unroll
    for (int t = 0; t < 8; ++t) O[base + 16 * t + l16] = oacc[t][v] * inv;
  }
}

// ---------------------------------------------------------------- launch
extern "C" void kernel_launch(void* const* d_in, const int* in_sizes, int n_in,
                              void* d_out, int out_size, void* d_ws, size_t ws_size,
                              hipStream_t stream) {
  const float* x    = (const float*)d_in[0];
  const float* Wq   = (const float*)d_in[1];
  const float* Wkvd = (const float*)d_in[2];
  const float* Wku  = (const float*)d_in[3];
  const float* Wvu  = (const float*)d_in[4];
  float* out = (float*)d_out;

  const size_t nx   = (size_t)BATCH * SEQ * EMBED;   // 16M
  const size_t nwq  = (size_t)EMBED * EMBED;         // 4M
  const size_t nwkd = (size_t)EMBED * LATENT;        // 1M
  const size_t nwup = (size_t)LATENT * EMBED;        // 1M
  const size_t nckv = (size_t)BATCH * SEQ * LATENT;  // 4M

  __bf16* ws    = (__bf16*)d_ws;
  __bf16* xb    = ws; ws += nx;
  __bf16* wqbT  = ws; ws += nwq;    // (N=E) x (K=E)
  __bf16* wkdbT = ws; ws += nwkd;   // (N=L) x (K=E)
  __bf16* wkubT = ws; ws += nwup;   // (N=E) x (K=L)
  __bf16* wvubT = ws; ws += nwup;   // (N=E) x (K=L)
  __bf16* qb    = ws; ws += nx;
  __bf16* ckv   = ws; ws += nckv;
  __bf16* kb    = ws; ws += nx;
  __bf16* vT    = ws; ws += nx;     // (B, E, S)

  {  // x: straight convert
    int n4 = (int)(nx / 4);
    cvt_f32_bf16<<<(n4 + 255) / 256, 256, 0, stream>>>(x, xb, n4);
  }
  // weights: convert + transpose (K x N fp32 -> N x K bf16)
  cvt_t_f32_bf16<<<dim3(EMBED / 32, EMBED / 32), 256, 0, stream>>>(Wq, wqbT, EMBED, EMBED);
  cvt_t_f32_bf16<<<dim3(EMBED / 32, LATENT / 32), 256, 0, stream>>>(Wkvd, wkdbT, EMBED, LATENT);
  cvt_t_f32_bf16<<<dim3(LATENT / 32, EMBED / 32), 256, 0, stream>>>(Wku, wkubT, LATENT, EMBED);
  cvt_t_f32_bf16<<<dim3(LATENT / 32, EMBED / 32), 256, 0, stream>>>(Wvu, wvubT, LATENT, EMBED);

  const int M = BATCH * SEQ;  // 8192
  dim3 blk(256);
  gemm_bf16_wmma<false><<<dim3(M / GTM, EMBED / GTN), blk, 0, stream>>>(xb, wqbT, qb, M, EMBED, EMBED);
  gemm_bf16_wmma<false><<<dim3(M / GTM, LATENT / GTN), blk, 0, stream>>>(xb, wkdbT, ckv, M, LATENT, EMBED);
  gemm_bf16_wmma<false><<<dim3(M / GTM, EMBED / GTN), blk, 0, stream>>>(ckv, wkubT, kb, M, EMBED, LATENT);
  gemm_bf16_wmma<true><<<dim3(M / GTM, EMBED / GTN), blk, 0, stream>>>(ckv, wvubT, vT, M, EMBED, LATENT);

  const int nrope = BATCH * SEQ * HEADS;
  rope_inplace<<<(nrope + 127) / 128, 128, 0, stream>>>(qb);
  rope_inplace<<<(nrope + 127) / 128, 128, 0, stream>>>(kb);

  const float scale = 0.08838834764831845f;  // 1/sqrt(128)
  mla_attention<<<dim3(BATCH * HEADS, SEQ / AQ), blk, 0, stream>>>(qb, kb, vT, out, scale);
}
